// WindowedSelfAttention_87832081203394
// MI455X (gfx1250) — matrix operand order repaired and checked by reading, not compiled, not run on hardware
//
#include <hip/hip_runtime.h>
#include <hip/hip_bf16.h>
#include <stddef.h>
#include <stdint.h>

// ---------------------------------------------------------------------------
// Types for CDNA5 WMMA
// ---------------------------------------------------------------------------
typedef __bf16 bf16_t;
typedef __attribute__((ext_vector_type(16))) __bf16 v16bf;
typedef __attribute__((ext_vector_type(8)))  __bf16 v8bf;
typedef __attribute__((ext_vector_type(4)))  __bf16 v4bf;
typedef __attribute__((ext_vector_type(8)))  float  v8f;
typedef __attribute__((ext_vector_type(4)))  float  v4f;

// Problem constants (match reference: B=2, T=2048, C=1024, H=16, D=64, WIN=16)
#define T_SEQ   2048
#define C_DIM   1024
#define N_HEAD  16
#define HEAD_D  64
#define WIN     16

// GEMM tiling
#define BM 128
#define BN 128
#define BK 32
#define LDA 40   // bf16 elems per LDS A row (32 K + 8 pad) -> 80B rows, 16B aligned
#define LDB 40   // same for B (stored [n][k])

// ---------------------------------------------------------------------------
// f32 -> bf16 elementwise convert (vectorized by 4)
// ---------------------------------------------------------------------------
__global__ __launch_bounds__(256)
void conv_bf16_kernel(const float* __restrict__ in, bf16_t* __restrict__ out) {
  int i = blockIdx.x * blockDim.x + threadIdx.x;
  v4f f = ((const v4f*)in)[i];
  v4bf o = { (bf16_t)f.x, (bf16_t)f.y, (bf16_t)f.z, (bf16_t)f.w };
  ((v4bf*)out)[i] = o;
}

// ---------------------------------------------------------------------------
// f32 [R,C] -> bf16 [C,R] tiled transpose-convert (64x64 tiles via LDS)
// ---------------------------------------------------------------------------
__global__ __launch_bounds__(256)
void transpose_conv_kernel(const float* __restrict__ in, bf16_t* __restrict__ out,
                           int R, int Cc) {
  __shared__ float tile[64][65];
  const int tr = blockIdx.y * 64;
  const int tc = blockIdx.x * 64;
  const int tid = threadIdx.x;
#pragma unroll
  for (int i = 0; i < 4; ++i) {
    int idx = tid + i * 256;            // 0..1023 float4 chunks
    int row = idx >> 4;                 // 0..63
    int c4  = (idx & 15) << 2;          // 0..60
    v4f f = *(const v4f*)(in + (size_t)(tr + row) * Cc + tc + c4);
    tile[row][c4 + 0] = f.x; tile[row][c4 + 1] = f.y;
    tile[row][c4 + 2] = f.z; tile[row][c4 + 3] = f.w;
  }
  __syncthreads();
#pragma unroll
  for (int i = 0; i < 4; ++i) {
    int idx = tid + i * 256;
    int orow = idx >> 4;                // 0..63 (output row = original col)
    int oc4  = (idx & 15) << 2;         // 0..60 (output col = original row)
    v4bf o = { (bf16_t)tile[oc4 + 0][orow], (bf16_t)tile[oc4 + 1][orow],
               (bf16_t)tile[oc4 + 2][orow], (bf16_t)tile[oc4 + 3][orow] };
    *(v4bf*)(out + (size_t)(tc + orow) * R + tr + oc4) = o;
  }
}

// ---------------------------------------------------------------------------
// bf16 WMMA GEMM with async global->LDS staging, double buffered.
//   C[M,N] (f32) = A[M,K] (bf16, row-major) @ Bt[N,K]^T (bf16, row-major [N,K])
// Requires M%128==0, N%128==0, K%32==0.
// ---------------------------------------------------------------------------
__global__ __launch_bounds__(256)
void gemm_bf16_wmma(const bf16_t* __restrict__ A, const bf16_t* __restrict__ Bt,
                    float* __restrict__ C, int M, int N, int K) {
  __shared__ bf16_t As[2][BM * LDA];   // As[buf][m][k]
  __shared__ bf16_t Bs[2][BN * LDB];   // Bs[buf][n][k]

  const int tid     = threadIdx.x;
  const int lane    = tid & 31;
  const int wave    = tid >> 5;            // 0..7
  const int wm      = (wave >> 2) * 64;    // 0 / 64
  const int wn      = (wave & 3)  * 32;    // 0/32/64/96
  const int l16     = lane & 15;
  const int halfsel = lane >> 4;

  const int block_m = blockIdx.y * BM;
  const int block_n = blockIdx.x * BN;

  // staging: tile = 128 rows x 64B = 512 x 16B chunks; 2 chunks/thread/matrix
  const int srow = tid >> 2;               // 0..63 (and +64 on second chunk)
  const int scoff = (tid & 3) << 4;        // byte offset in row: 0/16/32/48

  const uint64_t abase = (uint64_t)(uintptr_t)A;
  const uint64_t bbase = (uint64_t)(uintptr_t)Bt;

  v8f acc[4][2];
#pragma unroll
  for (int mi = 0; mi < 4; ++mi)
#pragma unroll
    for (int ni = 0; ni < 2; ++ni)
      acc[mi][ni] = (v8f){0.f, 0.f, 0.f, 0.f, 0.f, 0.f, 0.f, 0.f};

  const int ak0 = halfsel ? 8 : 0;         // A frag K chunk base
  const int bk0 = halfsel ? 8 : 0;         // B frag K chunk base

  const int nk = K / BK;

#define ISSUE_TILE(buf, k0)                                                       \
  {                                                                               \
    uint32_t albase = (uint32_t)(uintptr_t)(&As[(buf)][0]);                       \
    uint32_t blbase = (uint32_t)(uintptr_t)(&Bs[(buf)][0]);                       \
    _Pragma("unroll")                                                             \
    for (int i = 0; i < 2; ++i) {                                                 \
      int row = srow + i * 64;                                                    \
      uint64_t ga = abase + (((uint64_t)(block_m + row) * K + (k0)) << 1) + scoff;\
      uint32_t la = albase + row * (LDA * 2) + scoff;                             \
      asm volatile("global_load_async_to_lds_b128 %0, %1, off"                    \
                   :: "v"(la), "v"(ga) : "memory");                               \
      uint64_t gb = bbase + (((uint64_t)(block_n + row) * K + (k0)) << 1) + scoff;\
      uint32_t lb = blbase + row * (LDB * 2) + scoff;                             \
      asm volatile("global_load_async_to_lds_b128 %0, %1, off"                    \
                   :: "v"(lb), "v"(gb) : "memory");                               \
    }                                                                             \
  }

  ISSUE_TILE(0, 0)

  for (int kt = 0; kt < nk; ++kt) {
    const int cur = kt & 1;
    if (kt + 1 < nk) {
      ISSUE_TILE(cur ^ 1, (kt + 1) * BK)
      // 4 in-flight copies belong to the next tile; current tile is done at <=4
      asm volatile("s_wait_asynccnt 0x4" ::: "memory");
    } else {
      asm volatile("s_wait_asynccnt 0x0" ::: "memory");
    }
    __syncthreads();

    // ---- fragments per CDNA5 WMMA VGPR layouts ----
    v16bf afrag[4];
#pragma unroll
    for (int mi = 0; mi < 4; ++mi) {
      const bf16_t* ap = &As[cur][(wm + mi * 16 + l16) * LDA + ak0];
      v8bf alo = *(const v8bf*)(ap);                 // K = ak0 .. ak0+7
      v8bf ahi = *(const v8bf*)(ap + 16);            // K = ak0+16 .. ak0+23
      afrag[mi] = __builtin_shufflevector(alo, ahi,
                    0, 1, 2, 3, 4, 5, 6, 7, 8, 9, 10, 11, 12, 13, 14, 15);
    }
    v16bf bfrag[2];
#pragma unroll
    for (int ni = 0; ni < 2; ++ni) {
      const bf16_t* bp = &Bs[cur][(wn + ni * 16 + l16) * LDB + bk0];
      v8bf blo = *(const v8bf*)(bp);                 // K = bk0 .. bk0+7
      v8bf bhi = *(const v8bf*)(bp + 16);            // K = bk0+16 .. bk0+23
      bfrag[ni] = __builtin_shufflevector(blo, bhi,
                    0, 1, 2, 3, 4, 5, 6, 7, 8, 9, 10, 11, 12, 13, 14, 15);
    }

#pragma unroll
    for (int mi = 0; mi < 4; ++mi)
#pragma unroll
      for (int ni = 0; ni < 2; ++ni)
        acc[mi][ni] = __builtin_amdgcn_wmma_f32_16x16x32_bf16(
            false, afrag[mi], false, bfrag[ni],
            (short)0, acc[mi][ni], false, false);
    __syncthreads();   // protect buffer reuse (rewritten by next iteration's prefetch)
  }
#undef ISSUE_TILE

  // ---- epilogue: C/D layout (VGPR r -> row r + 8*halfsel, col = l16) ----
#pragma unroll
  for (int mi = 0; mi < 4; ++mi) {
#pragma unroll
    for (int ni = 0; ni < 2; ++ni) {
      const int col = block_n + wn + ni * 16 + l16;
      float* cp = C + (size_t)(block_m + wm + mi * 16 + halfsel * 8) * N + col;
#pragma unroll
      for (int r = 0; r < 8; ++r) {
        *cp = acc[mi][ni][r];
        cp += N;
      }
    }
  }
}

// ---------------------------------------------------------------------------
// Windowed attention: one thread per (b,h,t). qkv f32 [B*T, 3C]; output y is
// written directly in bf16 (it feeds the bf16 projection GEMM).
// ---------------------------------------------------------------------------
__global__ __launch_bounds__(256)
void win_attn_kernel(const float* __restrict__ qkv, bf16_t* __restrict__ y) {
  const int idx = blockIdx.x * blockDim.x + threadIdx.x;   // 0 .. B*H*T-1
  const int t  = idx & (T_SEQ - 1);
  const int bh = idx >> 11;                                // T = 2048 = 2^11
  const int h  = bh & (N_HEAD - 1);
  const int b  = bh >> 4;

  const size_t rowstride = 3 * C_DIM;
  const size_t btbase = (size_t)(b * T_SEQ + t) * rowstride;
  const int    hoff   = h * HEAD_D;

  float q[HEAD_D];
#pragma unroll
  for (int d = 0; d < HEAD_D; ++d) q[d] = qkv[btbase + hoff + d];

  const float scale = 0.125f;   // 1/sqrt(64)
  float sc[WIN];
  float mx = -3.0e38f;
#pragma unroll
  for (int j = 0; j < WIN; ++j) {
    int s = t - (WIN - 1) + j;
    bool valid = (s >= 0);
    int sl = valid ? s : 0;
    const float* kp = qkv + (size_t)(b * T_SEQ + sl) * rowstride + C_DIM + hoff;
    float d0 = 0.f;
#pragma unroll
    for (int d = 0; d < HEAD_D; ++d) d0 = fmaf(q[d], kp[d], d0);
    sc[j] = valid ? d0 * scale : -3.0e38f;
    mx = fmaxf(mx, sc[j]);
  }
  float sum = 0.f;
#pragma unroll
  for (int j = 0; j < WIN; ++j) {
    sc[j] = __expf(sc[j] - mx);   // masked entries underflow to 0
    sum += sc[j];
  }
  const float inv = 1.0f / sum;

  float o[HEAD_D];
#pragma unroll
  for (int d = 0; d < HEAD_D; ++d) o[d] = 0.f;
#pragma unroll
  for (int j = 0; j < WIN; ++j) {
    int s = t - (WIN - 1) + j;
    int sl = (s >= 0) ? s : 0;
    const float p = sc[j] * inv;
    const float* vp = qkv + (size_t)(b * T_SEQ + sl) * rowstride + 2 * C_DIM + hoff;
#pragma unroll
    for (int d = 0; d < HEAD_D; ++d) o[d] = fmaf(p, vp[d], o[d]);
  }

  bf16_t* yp = y + (size_t)(b * T_SEQ + t) * C_DIM + hoff;
#pragma unroll
  for (int d = 0; d < HEAD_D; ++d) yp[d] = (bf16_t)o[d];
}

// ---------------------------------------------------------------------------
// Launch: prep (convert/transpose) -> GEMM(qkv) -> attention -> GEMM(proj)
// ---------------------------------------------------------------------------
extern "C" void kernel_launch(void* const* d_in, const int* in_sizes, int n_in,
                              void* d_out, int out_size, void* d_ws, size_t ws_size,
                              hipStream_t stream) {
  const float* x      = (const float*)d_in[0];   // [B*T, C]
  const float* w_qkv  = (const float*)d_in[1];   // [C, 3C]
  const float* w_proj = (const float*)d_in[2];   // [C, C]
  float*       out    = (float*)d_out;           // [B*T, C]

  const int M  = in_sizes[0] / C_DIM;            // B*T = 4096
  const int N1 = 3 * C_DIM;                      // 3072
  const int N2 = C_DIM;                          // 1024
  const int K  = C_DIM;                          // 1024

  // workspace layout
  char* ws = (char*)d_ws;
  float*  qkv    = (float*)(ws);                                   // M*3C f32 = 48MB
  bf16_t* x_bf   = (bf16_t*)(ws + (size_t)M * N1 * 4);             // M*C bf16 = 8MB
  bf16_t* wqkvT  = (bf16_t*)(ws + (size_t)M * N1 * 4
                                + (size_t)M * K * 2);              // [3C, C] = 6MB
  bf16_t* wprojT = (bf16_t*)((char*)wqkvT + (size_t)N1 * K * 2);   // [C, C]  = 2MB
  bf16_t* y_bf   = (bf16_t*)((char*)wprojT + (size_t)N2 * K * 2);  // M*C bf16 = 8MB

  dim3 blk(256);

  // 0a) x -> bf16
  conv_bf16_kernel<<<(M * K / 4) / 256, blk, 0, stream>>>(x, x_bf);
  // 0b) w_qkv [K,3C] -> wqkvT [3C,K] bf16 ; w_proj [K,C] -> wprojT [C,K] bf16
  transpose_conv_kernel<<<dim3(N1 / 64, K / 64), blk, 0, stream>>>(w_qkv, wqkvT, K, N1);
  transpose_conv_kernel<<<dim3(N2 / 64, K / 64), blk, 0, stream>>>(w_proj, wprojT, K, N2);

  // 1) qkv = x @ w_qkv
  gemm_bf16_wmma<<<dim3(N1 / BN, M / BM), blk, 0, stream>>>(x_bf, wqkvT, qkv, M, N1, K);

  // 2) windowed attention (writes y in bf16)
  win_attn_kernel<<<(M * N_HEAD) / 256, blk, 0, stream>>>(qkv, y_bf);

  // 3) out = y @ w_proj
  gemm_bf16_wmma<<<dim3(N2 / BN, M / BM), blk, 0, stream>>>(y_bf, wprojT, out, M, N2, K);
}
